// SineLSTM_8246337208791
// MI455X (gfx1250) — matrix operand order repaired and checked by reading, not compiled, not run on hardware
//
#include <hip/hip_runtime.h>

typedef __attribute__((ext_vector_type(16))) _Float16 v16h;
typedef __attribute__((ext_vector_type(8)))  float    v8f;

#define HSZ   50
#define G4    200
#define NT    16            // N tiles padded 13 -> 16: every wave owns exactly 2, branch-free
#define T_LEN 1024
#define PRED  32
#define OUTW  (T_LEN + PRED)
#define MB    16            // batch rows per block
#define THREADS 256
#define WAVES 8
#define TPW   2             // tiles per wave (16 / 8)
#define NCHUNK (T_LEN / 64)

// Inverse of this file's contiguous-K 16-bit A-fragment layout: element (row M, K) -> flat half index.
// lanes 0-15 hold K = {0..15}; lanes 16-31 hold K = {16..31} (per 32-K fragment)
__device__ __forceinline__ int af_index(int row, int K) {
    int kk = K >> 5;
    int Kl = K & 31;
    int l  = row + 16 * ((Kl >> 4) & 1);
    int j  = Kl & 15;
    return (kk * 32 + l) * 16 + j;
}
// Fragment-order gates buffer [NT][32 lanes][8 floats]: element (row, col) -> float index.
// C/D layout: VGPR v, lanes 0-15: M=v, N=lane; lanes 16-31: M=v+8, N=lane-16.
__device__ __forceinline__ int gate_off(int row, int col) {
    int l = (col & 15) + 16 * (row >> 3);
    return ((col >> 4) * 32 + l) * 8 + (row & 7);
}

// gfx1250 native V_TANH_F32; sigmoid = 0.5*tanh(0.5x)+0.5 -> one transcendental, no division.
#if __has_builtin(__builtin_amdgcn_tanhf)
__device__ __forceinline__ float tanh_hw(float v) { return __builtin_amdgcn_tanhf(v); }
#else
__device__ __forceinline__ float tanh_hw(float v) {
    return 2.f * __builtin_amdgcn_rcpf(1.f + __expf(-2.f * v)) - 1.f;
}
#endif
__device__ __forceinline__ float sigmoid_hw(float v) { return 0.5f * tanh_hw(0.5f * v) + 0.5f; }

// Build a B-fragment (16x16x32 f16) in registers straight from global memory.
// B layout: lane l -> column n*16+(l&15); halves j -> K = kk*32 + 16*(l>>4) + j (contiguous in W).
// Row K==50 carries the (optional) bias, paired with the constant 1.0 planted at K==50 of A.
// Fast path (nv>=16) is unguarded + contiguous -> vectorizable global loads; the bias row
// can only fall in the Kb=48 tail block, where K==50 <=> j==nv.
__device__ __forceinline__ v16h load_bfrag(const float* __restrict__ Wg,
                                           const float* __restrict__ ba,
                                           const float* __restrict__ bb,
                                           int n, int kk, int lane) {
    v16h r;
    int col = n * 16 + (lane & 15);
    int Kb  = kk * 32 + ((lane >> 4) << 4);   // 0,16,32,48
    int nv  = HSZ - Kb;                       // 50,34,18,2
    const float* src = Wg + col * HSZ + Kb;
    if (col < G4) {
        if (nv >= 16) {
#pragma unroll
            for (int j = 0; j < 16; ++j) r[j] = (_Float16)src[j];
        } else {                               // Kb=48 tail: j<nv weights, j==nv bias, else 0
            float bv = ba ? (ba[col] + bb[col]) : 0.f;
#pragma unroll
            for (int j = 0; j < 16; ++j) {
                float v = (j < nv) ? src[j] : ((j == nv) ? bv : 0.f);
                r[j] = (_Float16)v;
            }
        }
    } else {
#pragma unroll
        for (int j = 0; j < 16; ++j) r[j] = (_Float16)0.f;
    }
    return r;
}

__global__ __launch_bounds__(THREADS)
void sine_lstm_kernel(const float* __restrict__ x,
                      const float* __restrict__ Wih1, const float* __restrict__ Whh1,
                      const float* __restrict__ bih1, const float* __restrict__ bhh1,
                      const float* __restrict__ Wih2, const float* __restrict__ Whh2,
                      const float* __restrict__ bih2, const float* __restrict__ bhh2,
                      const float* __restrict__ Wlin, const float* __restrict__ blin,
                      float* __restrict__ out)
{
    __shared__ __align__(32) float    gates[NT * 32 * 8];    // 16 KB, fragment order
    __shared__ __align__(32) _Float16 AF1[2 * 32 * 16];      // h1 A-fragments (+1.0 at K=50)
    __shared__ __align__(32) _Float16 AF2[2 * 32 * 16];      // h2 A-fragments (+1.0 at K=50)
    __shared__ __align__(32) float    part[MB * 16];         // linear-head partials
    __shared__ __align__(32) float    xst[2 * 64 * MB];      // double-buffered x chunks
    __shared__ __align__(32) float    lout[MB];              // last output (predict feedback)

    const int tid   = threadIdx.x;
    const int lane  = tid & 31;
    const int wave  = tid >> 5;
    const int rHalf = lane >> 4;
    const int nc    = lane & 15;
    const int rowBase = blockIdx.x * MB;
    const int n0 = wave * TPW;                   // exactly 2 N-tiles per wave

    // ---------- prologue: weights (+fused biases) -> registers ----------
    v16h wb1[TPW][2], wb2i[TPW][2], wb2h[TPW][2];
    float wi1v[TPW];
#pragma unroll
    for (int i = 0; i < TPW; ++i) {
        int n = n0 + i;
#pragma unroll
        for (int kk = 0; kk < 2; ++kk) {
            wb1[i][kk]  = load_bfrag(Whh1, bih1, bhh1, n, kk, lane);  // bias1 in row 50
            wb2i[i][kk] = load_bfrag(Wih2, bih2, bhh2, n, kk, lane);  // bias2 in row 50
            wb2h[i][kk] = load_bfrag(Whh2, nullptr, nullptr, n, kk, lane);
        }
        int col = n * 16 + nc;
        wi1v[i] = (col < G4) ? Wih1[col] : 0.f;
    }

    // ---------- precomputed activation indexing (identical every step) ----------
    const int arow = tid & 15;          // batch row in activation phases
    const int hb   = (tid >> 4) & 15;   // 0..15 (masked -> provable range -> branchless it 0..2)
    int gOff[4][4], aOff[4];
    float wlA[4];
#pragma unroll
    for (int it = 0; it < 4; ++it) {
        int hh = hb + 16 * it;
        int hc = (hh < HSZ) ? hh : 0;
#pragma unroll
        for (int g = 0; g < 4; ++g) gOff[it][g] = gate_off(arow, g * HSZ + hc);
        aOff[it] = af_index(arow, hc);
        wlA[it]  = (hh < HSZ) ? Wlin[hh] : 0.f;   // linear-head tap for this state
    }
    const float blv = blin[0];

    // cell states live in registers: the (arow, hh) -> lane mapping is static for all steps
    float c1r[4] = { 0.f, 0.f, 0.f, 0.f };
    float c2r[4] = { 0.f, 0.f, 0.f, 0.f };

    // ---------- zero initial hidden state; plant the bias-ones at K=50 ----------
    for (int i = tid; i < 2 * 32 * 16; i += THREADS) { AF1[i] = (_Float16)0.f; AF2[i] = (_Float16)0.f; }
    __syncthreads();
    if (tid < MB) {
        AF1[af_index(tid, HSZ)] = (_Float16)1.f;   // activations only write K<50: this persists
        AF2[af_index(tid, HSZ)] = (_Float16)1.f;
        lout[tid] = 0.f;
    }
    __syncthreads();

    // ---------- activation: gate nonlinearities + state update (+ fused linear partial) ----------
    auto act_one = [&](float* cr, _Float16* AF, int it) -> float {
        float gi = gates[gOff[it][0]];
        float gf = gates[gOff[it][1]];
        float gg = gates[gOff[it][2]];
        float go = gates[gOff[it][3]];
        float cn = sigmoid_hw(gf) * cr[it] + sigmoid_hw(gi) * tanh_hw(gg);
        cr[it] = cn;
        float hn = sigmoid_hw(go) * tanh_hw(cn);
        AF[aOff[it]] = (_Float16)hn;
        return hn;
    };
    auto act = [&](float* cr, _Float16* AF, bool withPartial) {
        float p = 0.f;
#pragma unroll
        for (int it = 0; it < 3; ++it) {              // hh = hb+16*it <= 47: always valid
            float h = act_one(cr, AF, it);
            p += h * wlA[it];
        }
        if (hb < 2) {                                 // hh = hb+48 < 50 tail
            float h = act_one(cr, AF, 3);
            p += h * wlA[3];
        }
        if (withPartial) part[arow * 16 + hb] = p;    // dead code for layer 1 (p unused)
    };

    // ---------- one recurrent step: 4 barriers, no trailing barrier ----------
    auto step = [&](const float* __restrict__ xv, int ocol) {
        // ---- layer 1: gates = x*Wih1 + [h1|1] @ [Whh1|b1]^T ----
        {
            float4 xa = *(const float4*)(xv + rHalf * 8);
            float4 xb = *(const float4*)(xv + rHalf * 8 + 4);
            float xr[8] = { xa.x, xa.y, xa.z, xa.w, xb.x, xb.y, xb.z, xb.w };
            v16h a0 = *(const v16h*)(AF1 + lane * 16);
            v16h a1 = *(const v16h*)(AF1 + (32 + lane) * 16);
#pragma unroll
            for (int i = 0; i < TPW; ++i) {
                v8f c;
#pragma unroll
                for (int v = 0; v < 8; ++v) c[v] = xr[v] * wi1v[i];
                c = __builtin_amdgcn_wmma_f32_16x16x32_f16(false, a0, false, wb1[i][0],
                                                           (short)0, c, false, false);
                c = __builtin_amdgcn_wmma_f32_16x16x32_f16(false, a1, false, wb1[i][1],
                                                           (short)0, c, false, false);
                *(v8f*)(gates + ((n0 + i) * 32 + lane) * 8) = c;
            }
        }
        __syncthreads();
        act(c1r, AF1, false);
        __syncthreads();
        // ---- layer 2: gates = [h1|1] @ [Wih2|b2]^T + [h2|1] @ [Whh2|0]^T ----
        {
            v16h a0 = *(const v16h*)(AF1 + lane * 16);
            v16h a1 = *(const v16h*)(AF1 + (32 + lane) * 16);
            v16h a2 = *(const v16h*)(AF2 + lane * 16);
            v16h a3 = *(const v16h*)(AF2 + (32 + lane) * 16);
#pragma unroll
            for (int i = 0; i < TPW; ++i) {
                v8f c = {};    // inline-0 C operand for the first WMMA
                c = __builtin_amdgcn_wmma_f32_16x16x32_f16(false, a0, false, wb2i[i][0],
                                                           (short)0, c, false, false);
                c = __builtin_amdgcn_wmma_f32_16x16x32_f16(false, a1, false, wb2i[i][1],
                                                           (short)0, c, false, false);
                c = __builtin_amdgcn_wmma_f32_16x16x32_f16(false, a2, false, wb2h[i][0],
                                                           (short)0, c, false, false);
                c = __builtin_amdgcn_wmma_f32_16x16x32_f16(false, a3, false, wb2h[i][1],
                                                           (short)0, c, false, false);
                *(v8f*)(gates + ((n0 + i) * 32 + lane) * 8) = c;
            }
        }
        __syncthreads();
        act(c2r, AF2, true);     // also produces this lane's linear-head partial
        __syncthreads();
        // ---- linear head: 16-lane reduction + output ----
        if (tid < MB) {
            const float4* pp = (const float4*)(part + tid * 16);
            float4 s0 = pp[0], s1 = pp[1], s2 = pp[2], s3 = pp[3];
            float acc = blv
                + s0.x + s0.y + s0.z + s0.w
                + s1.x + s1.y + s1.z + s1.w
                + s2.x + s2.y + s2.z + s2.w
                + s3.x + s3.y + s3.z + s3.w;
            out[(size_t)(rowBase + tid) * OUTW + ocol] = acc;
            lout[tid] = acc;
        }
    };

    // ---------- main scan over T: double-buffered x chunks, prefetch next chunk ----------
    const int srow = tid >> 4;            // 0..15
    const int stt  = (tid & 15) * 4;      // 0..60
    const float* xsrc = x + (size_t)(rowBase + srow) * T_LEN + stt;

    float4 xpre = *(const float4*)(xsrc);                     // chunk 0
    for (int c = 0; c < NCHUNK; ++c) {
        float* dst = xst + (c & 1) * (64 * MB);
        dst[(stt + 0) * MB + srow] = xpre.x;
        dst[(stt + 1) * MB + srow] = xpre.y;
        dst[(stt + 2) * MB + srow] = xpre.z;
        dst[(stt + 3) * MB + srow] = xpre.w;
        __syncthreads();
        if (c + 1 < NCHUNK)                                   // prefetch next chunk
            xpre = *(const float4*)(xsrc + (c + 1) * 64);
        for (int tt = 0; tt < 64; ++tt)
            step(dst + tt * MB, c * 64 + tt);
    }
    // ---------- autoregressive predict: feed output back as x ----------
    for (int p = 0; p < PRED; ++p) {
        __syncthreads();                  // protect lout (written by reduction, read as x here)
        step(lout, T_LEN + p);
    }
}

extern "C" void kernel_launch(void* const* d_in, const int* in_sizes, int n_in,
                              void* d_out, int out_size, void* d_ws, size_t ws_size,
                              hipStream_t stream) {
    const float* x    = (const float*)d_in[0];
    const float* Wih1 = (const float*)d_in[1];
    const float* Whh1 = (const float*)d_in[2];
    const float* bih1 = (const float*)d_in[3];
    const float* bhh1 = (const float*)d_in[4];
    const float* Wih2 = (const float*)d_in[5];
    const float* Whh2 = (const float*)d_in[6];
    const float* bih2 = (const float*)d_in[7];
    const float* bhh2 = (const float*)d_in[8];
    const float* Wlin = (const float*)d_in[9];
    const float* blin = (const float*)d_in[10];
    float* out = (float*)d_out;

    int B = in_sizes[0] / T_LEN;      // 512
    int grid = B / MB;                // 32 blocks, each owns 16 independent sequences

    sine_lstm_kernel<<<dim3(grid), dim3(THREADS), 0, stream>>>(
        x, Wih1, Whh1, bih1, bhh1, Wih2, Whh2, bih2, bhh2, Wlin, blin, out);
}
